// CodeBook_76682346102955
// MI455X (gfx1250) — compile-verified, matrix operand order checked by reading, not run on hardware
//
#include <hip/hip_runtime.h>
#include <math.h>
#include <stdint.h>

typedef __attribute__((ext_vector_type(16))) _Float16 v16h;
typedef __attribute__((ext_vector_type(8)))  _Float16 h8;
typedef __attribute__((ext_vector_type(8)))  float    v8f;

#define N_IN   32768
#define K_EMB  8192
#define DIM    512
#define CCOST  0.25f

#define BM 128          // inputs rows per block
#define BN 128          // codebook rows per block
#define BK 32           // D-chunk (WMMA K for f16)
#define LDA 40          // padded LDS row stride in halves (80B -> spreads banks, keeps 16B align)
#define NCH (DIM / BK)  // 16 chunks

// ---------------- row norms ----------------
__global__ __launch_bounds__(128)
void row_norm_kernel(const float* __restrict__ src, float* __restrict__ dst, int nrows) {
    __shared__ float red[128];
    const int row = blockIdx.x;
    if (row >= nrows) return;
    const float* p = src + (size_t)row * DIM;
    float s = 0.f;
    for (int i = threadIdx.x; i < DIM; i += 128) { float v = p[i]; s += v * v; }
    red[threadIdx.x] = s;
    __syncthreads();
    for (int off = 64; off > 0; off >>= 1) {
        if (threadIdx.x < off) red[threadIdx.x] += red[threadIdx.x + off];
        __syncthreads();
    }
    if (threadIdx.x == 0) dst[row] = red[0];
}

// ---------------- init accumulators ----------------
__global__ void init_kernel(float* __restrict__ counts, float* __restrict__ loss_accum) {
    const int i = blockIdx.x * 256 + threadIdx.x;
    if (i < K_EMB) counts[i] = 0.f;
    if (i == 0) *loss_accum = 0.f;
}

// ---------------- f32 -> f16 pre-conversion (fast path) ----------------
__device__ __forceinline__ h8 pack_h8(const float4 a, const float4 b) {
    h8 r;
    r[0] = (_Float16)a.x; r[1] = (_Float16)a.y; r[2] = (_Float16)a.z; r[3] = (_Float16)a.w;
    r[4] = (_Float16)b.x; r[5] = (_Float16)b.y; r[6] = (_Float16)b.z; r[7] = (_Float16)b.w;
    return r;
}

__global__ __launch_bounds__(256)
void to_f16_kernel(const float* __restrict__ src, _Float16* __restrict__ dst, int n8) {
    const int i = blockIdx.x * 256 + threadIdx.x;   // 8 elements per thread
    if (i < n8) {
        const float4* s = (const float4*)src + (size_t)i * 2;
        float4 x0 = s[0], x1 = s[1];
        *(h8*)(dst + (size_t)i * 8) = pack_h8(x0, x1);
    }
}

// ---------------- common WMMA fragment/epilogue helpers ----------------
union F16Frag { v16h v; h8 h[2]; };

// copy 32B of A-tile and 32B of B-tile per thread via the CDNA5 async LDS path
// (ASYNCcnt-tracked, bypasses the register file; INST_OFFSET applies to both
//  the LDS and global address per ISA 08_async_tensor §4.4)
__device__ __forceinline__ void async_copy_tile(uint32_t ldsA, uint32_t ldsB,
                                                const _Float16* ga, const _Float16* gb) {
    asm volatile(
        "global_load_async_to_lds_b128 %0, %2, off\n\t"
        "global_load_async_to_lds_b128 %0, %2, off offset:16\n\t"
        "global_load_async_to_lds_b128 %1, %3, off\n\t"
        "global_load_async_to_lds_b128 %1, %3, off offset:16"
        :: "v"(ldsA), "v"(ldsB), "v"(ga), "v"(gb) : "memory");
}

// ---------------- WMMA distance GEMM, async double-buffered f16 path ----------------
// dist[n,k] = ||x_n||^2 + ||w_k||^2 - 2 * (x_n . w_k)
__global__ __launch_bounds__(256)
void dist_gemm_async_kernel(const _Float16* __restrict__ Xh, const _Float16* __restrict__ Wh,
                            const float* __restrict__ xnorm, const float* __restrict__ wnorm,
                            float* __restrict__ dist) {
    __shared__ __align__(16) _Float16 lA[2][BM * LDA];
    __shared__ __align__(16) _Float16 lB[2][BN * LDA];

    const int tid  = threadIdx.x;
    const int wave = tid >> 5;
    const int lane = tid & 31;
    const int lr   = lane & 15;
    const int hi   = lane >> 4;
    const int waveM = wave >> 2;    // 0..1
    const int waveN = wave & 3;     // 0..3
    const int m0 = blockIdx.y * BM;
    const int k0 = blockIdx.x * BN;

    const int lrow = tid >> 1;          // 0..127
    const int lcol = (tid & 1) * 16;    // 0 or 16

    // per-thread source pointers for the 32B segment this thread stages
    const _Float16* ga = Xh + (size_t)(m0 + lrow) * DIM + lcol;
    const _Float16* gb = Wh + (size_t)(k0 + lrow) * DIM + lcol;
    // LDS byte offsets: flat LDS addressing keeps the offset in addr[31:0]
    const uint32_t lAoff[2] = { (uint32_t)(uintptr_t)&lA[0][lrow * LDA + lcol],
                                (uint32_t)(uintptr_t)&lA[1][lrow * LDA + lcol] };
    const uint32_t lBoff[2] = { (uint32_t)(uintptr_t)&lB[0][lrow * LDA + lcol],
                                (uint32_t)(uintptr_t)&lB[1][lrow * LDA + lcol] };

    v8f acc[4][2];
    const v8f vz = {0.f,0.f,0.f,0.f,0.f,0.f,0.f,0.f};
    #pragma unroll
    for (int i = 0; i < 4; ++i)
        #pragma unroll
        for (int j = 0; j < 2; ++j) acc[i][j] = vz;

    // prologue: chunk 0 -> buffer 0
    async_copy_tile(lAoff[0], lBoff[0], ga, gb);

    #pragma unroll
    for (int c = 0; c < NCH; ++c) {
        if (c + 1 < NCH) {
            async_copy_tile(lAoff[(c + 1) & 1], lBoff[(c + 1) & 1],
                            ga + (c + 1) * BK, gb + (c + 1) * BK);
            // 4 async ops/chunk in order: <=4 outstanding means chunk c landed
            asm volatile("s_wait_asynccnt 0x4" ::: "memory");
        } else {
            asm volatile("s_wait_asynccnt 0x0" ::: "memory");
        }
        __syncthreads();                       // all waves' chunk-c data in LDS

        const int sel = c & 1;
        F16Frag af[4], bf[2];
        #pragma unroll
        for (int i = 0; i < 4; ++i) {
            const int base = (waveM * 64 + i * 16 + lr) * LDA + hi * 8;
            af[i].h[0] = *(const h8*)&lA[sel][base];
            af[i].h[1] = *(const h8*)&lA[sel][base + 16];
        }
        #pragma unroll
        for (int j = 0; j < 2; ++j) {
            const int base = (waveN * 32 + j * 16 + lr) * LDA + hi * 8;
            bf[j].h[0] = *(const h8*)&lB[sel][base];
            bf[j].h[1] = *(const h8*)&lB[sel][base + 16];
        }
        #pragma unroll
        for (int i = 0; i < 4; ++i)
            #pragma unroll
            for (int j = 0; j < 2; ++j)
                acc[i][j] = __builtin_amdgcn_wmma_f32_16x16x32_f16(
                    false, af[i].v, false, bf[j].v, (short)0, acc[i][j], false, false);

        __syncthreads();   // buffer 'sel' free before next iteration's async issue
    }

    #pragma unroll
    for (int j = 0; j < 2; ++j) {
        const int gk = k0 + waveN * 32 + j * 16 + lr;
        const float wn = wnorm[gk];
        #pragma unroll
        for (int i = 0; i < 4; ++i) {
            const int mbase = m0 + waveM * 64 + i * 16 + hi * 8;
            #pragma unroll
            for (int e = 0; e < 8; ++e) {
                const int gm = mbase + e;
                dist[(size_t)gm * K_EMB + gk] = xnorm[gm] + wn - 2.f * acc[i][j][e];
            }
        }
    }
}

// ---------------- WMMA distance GEMM, f32-source fallback (small ws) ----------------
__global__ __launch_bounds__(256)
void dist_gemm_kernel(const float* __restrict__ X, const float* __restrict__ W,
                      const float* __restrict__ xnorm, const float* __restrict__ wnorm,
                      float* __restrict__ dist) {
    __shared__ __align__(16) _Float16 lA[BM * LDA];
    __shared__ __align__(16) _Float16 lB[BN * LDA];

    const int tid  = threadIdx.x;
    const int wave = tid >> 5;
    const int lane = tid & 31;
    const int lr   = lane & 15;
    const int hi   = lane >> 4;
    const int waveM = wave >> 2;
    const int waveN = wave & 3;
    const int m0 = blockIdx.y * BM;
    const int k0 = blockIdx.x * BN;

    const int lrow = tid >> 1;
    const int lcol = (tid & 1) * 16;

    v8f acc[4][2];
    const v8f vz = {0.f,0.f,0.f,0.f,0.f,0.f,0.f,0.f};
    #pragma unroll
    for (int i = 0; i < 4; ++i)
        #pragma unroll
        for (int j = 0; j < 2; ++j) acc[i][j] = vz;

    for (int d = 0; d < DIM; d += BK) {
        const float* gaf = X + (size_t)(m0 + lrow) * DIM + d + lcol;
        const float* gbf = W + (size_t)(k0 + lrow) * DIM + d + lcol;
        const float4* ga4 = (const float4*)gaf;
        const float4* gb4 = (const float4*)gbf;
        float4 a0 = ga4[0], a1 = ga4[1], a2 = ga4[2], a3 = ga4[3];
        float4 b0 = gb4[0], b1 = gb4[1], b2 = gb4[2], b3 = gb4[3];
        if (d + BK < DIM) {
            __builtin_prefetch(gaf + BK, 0, 0);
            __builtin_prefetch(gbf + BK, 0, 0);
        }
        __syncthreads();
        *(h8*)&lA[lrow * LDA + lcol]     = pack_h8(a0, a1);
        *(h8*)&lA[lrow * LDA + lcol + 8] = pack_h8(a2, a3);
        *(h8*)&lB[lrow * LDA + lcol]     = pack_h8(b0, b1);
        *(h8*)&lB[lrow * LDA + lcol + 8] = pack_h8(b2, b3);
        __syncthreads();

        F16Frag af[4], bf[2];
        #pragma unroll
        for (int i = 0; i < 4; ++i) {
            const int base = (waveM * 64 + i * 16 + lr) * LDA + hi * 8;
            af[i].h[0] = *(const h8*)&lA[base];
            af[i].h[1] = *(const h8*)&lA[base + 16];
        }
        #pragma unroll
        for (int j = 0; j < 2; ++j) {
            const int base = (waveN * 32 + j * 16 + lr) * LDA + hi * 8;
            bf[j].h[0] = *(const h8*)&lB[base];
            bf[j].h[1] = *(const h8*)&lB[base + 16];
        }
        #pragma unroll
        for (int i = 0; i < 4; ++i)
            #pragma unroll
            for (int j = 0; j < 2; ++j)
                acc[i][j] = __builtin_amdgcn_wmma_f32_16x16x32_f16(
                    false, af[i].v, false, bf[j].v, (short)0, acc[i][j], false, false);
    }

    #pragma unroll
    for (int j = 0; j < 2; ++j) {
        const int gk = k0 + waveN * 32 + j * 16 + lr;
        const float wn = wnorm[gk];
        #pragma unroll
        for (int i = 0; i < 4; ++i) {
            const int mbase = m0 + waveM * 64 + i * 16 + hi * 8;
            #pragma unroll
            for (int e = 0; e < 8; ++e) {
                const int gm = mbase + e;
                dist[(size_t)gm * K_EMB + gk] = xnorm[gm] + wn - 2.f * acc[i][j][e];
            }
        }
    }
}

// ---------------- fused row softmax(-d) + argmin ----------------
__global__ __launch_bounds__(256)
void softmax_argmin_kernel(float* __restrict__ probs, int* __restrict__ idx_out,
                           float* __restrict__ counts) {
    __shared__ float smin[256];
    __shared__ int   sidx[256];
    __shared__ float ssum[256];
    const int n = blockIdx.x;
    const int tid = threadIdx.x;
    float* row = probs + (size_t)n * K_EMB;

    float vals[32];                 // whole row in registers: 1 read + 1 write of the 1GB matrix
    float lmin = 3.402823466e38f;
    int   limin = 0;
    #pragma unroll
    for (int j = 0; j < 32; ++j) {
        const int c = tid + j * 256;
        const float v = row[c];
        vals[j] = v;
        if (v < lmin) { lmin = v; limin = c; }
    }
    smin[tid] = lmin; sidx[tid] = limin;
    __syncthreads();
    for (int off = 128; off > 0; off >>= 1) {
        if (tid < off && smin[tid + off] < smin[tid]) {
            smin[tid] = smin[tid + off]; sidx[tid] = sidx[tid + off];
        }
        __syncthreads();
    }
    const float dmin = smin[0];

    float ls = 0.f;
    #pragma unroll
    for (int j = 0; j < 32; ++j) ls += expf(dmin - vals[j]);
    ssum[tid] = ls;
    __syncthreads();
    for (int off = 128; off > 0; off >>= 1) {
        if (tid < off) ssum[tid] += ssum[tid + off];
        __syncthreads();
    }
    const float inv = 1.f / ssum[0];
    #pragma unroll
    for (int j = 0; j < 32; ++j)
        row[tid + j * 256] = expf(dmin - vals[j]) * inv;

    if (tid == 0) {
        idx_out[n] = sidx[0];
        atomicAdd(&counts[sidx[0]], 1.0f);
    }
}

// ---------------- gather quantized + loss partials ----------------
__global__ __launch_bounds__(256)
void gather_loss_kernel(const float* __restrict__ X, const float* __restrict__ W,
                        const int* __restrict__ idx, float* __restrict__ quant,
                        float* __restrict__ loss_accum) {
    __shared__ float red[256];
    const int n = blockIdx.x;
    const int tid = threadIdx.x;
    const int k = idx[n];
    const float* w = W + (size_t)k * DIM;
    const float* x = X + (size_t)n * DIM;
    float* q = quant + (size_t)n * DIM;
    float s = 0.f;
    for (int i = tid; i < DIM; i += 256) {
        const float wv = w[i], xv = x[i];
        const float df = wv - xv;
        q[i] = xv + df;             // straight-through: x + (q - x)
        s += df * df;
    }
    red[tid] = s;
    __syncthreads();
    for (int off = 128; off > 0; off >>= 1) {
        if (tid < off) red[tid] += red[tid + off];
        __syncthreads();
    }
    if (tid == 0) atomicAdd(loss_accum, red[0]);
}

// ---------------- scalars: loss + perplexity ----------------
__global__ __launch_bounds__(256)
void finalize_kernel(const float* __restrict__ counts, const float* __restrict__ loss_accum,
                     float* __restrict__ tail) {
    __shared__ float red[256];
    const int tid = threadIdx.x;
    float s = 0.f;
    for (int i = tid; i < K_EMB; i += 256) {
        const float avg = counts[i] * (1.0f / (float)N_IN);
        s += avg * logf(avg + 1e-10f);
    }
    red[tid] = s;
    __syncthreads();
    for (int off = 128; off > 0; off >>= 1) {
        if (tid < off) red[tid] += red[tid + off];
        __syncthreads();
    }
    if (tid == 0) {
        tail[0] = (1.0f + CCOST) * loss_accum[0] * (1.0f / 16777216.0f); // / (N*D)
        tail[1] = expf(-red[0]);
    }
}

extern "C" void kernel_launch(void* const* d_in, const int* in_sizes, int n_in,
                              void* d_out, int out_size, void* d_ws, size_t ws_size,
                              hipStream_t stream) {
    const float* X = (const float*)d_in[0];   // [N_IN, DIM]
    const float* W = (const float*)d_in[1];   // [K_EMB, DIM]

    float* out   = (float*)d_out;
    float* quant = out;                                   // N*DIM
    float* probs = out + (size_t)N_IN * DIM;              // N*K (dist written here first)
    float* tail  = probs + (size_t)N_IN * K_EMB;          // [loss, perplexity]

    float* xnorm   = (float*)d_ws;                        // N
    float* wnorm   = xnorm + N_IN;                        // K
    int*   idx     = (int*)(wnorm + K_EMB);               // N
    float* counts  = (float*)(idx + N_IN);                // K
    float* lossacc = counts + K_EMB;                      // 1

    const size_t small_bytes = (size_t)(N_IN + K_EMB + N_IN + K_EMB + 1) * 4;
    const size_t f16_off     = (small_bytes + 255) & ~(size_t)255;
    const size_t need        = f16_off + (size_t)(N_IN + K_EMB) * DIM * sizeof(_Float16);

    row_norm_kernel<<<N_IN, 128, 0, stream>>>(X, xnorm, N_IN);
    row_norm_kernel<<<K_EMB, 128, 0, stream>>>(W, wnorm, K_EMB);
    init_kernel<<<(K_EMB + 255) / 256, 256, 0, stream>>>(counts, lossacc);

    dim3 grid(K_EMB / BN, N_IN / BM);                     // 64 x 256 tiles
    if (ws_size >= need) {
        // fast path: one-time f16 conversion (120MB HBM, ~5us), then async-to-LDS GEMM
        _Float16* Xh = (_Float16*)((char*)d_ws + f16_off);
        _Float16* Wh = Xh + (size_t)N_IN * DIM;
        to_f16_kernel<<<(N_IN * DIM / 8 + 255) / 256, 256, 0, stream>>>(X, Xh, N_IN * DIM / 8);
        to_f16_kernel<<<(K_EMB * DIM / 8 + 255) / 256, 256, 0, stream>>>(W, Wh, K_EMB * DIM / 8);
        dist_gemm_async_kernel<<<grid, 256, 0, stream>>>(Xh, Wh, xnorm, wnorm, probs);
    } else {
        dist_gemm_kernel<<<grid, 256, 0, stream>>>(X, W, xnorm, wnorm, probs);
    }

    softmax_argmin_kernel<<<N_IN, 256, 0, stream>>>(probs, idx, counts);
    gather_loss_kernel<<<N_IN, 256, 0, stream>>>(X, W, idx, quant, lossacc);
    finalize_kernel<<<1, 256, 0, stream>>>(counts, lossacc, tail);
}
